// MPNNEncoder_57939108823255
// MI455X (gfx1250) — compile-verified, matrix operand order checked by reading, not run on hardware
//
#include <hip/hip_runtime.h>
#include <hip/hip_bf16.h>

#define NNODES 50000
#define NEDGES 800000
#define NGRAPH 64
#define DIN    128
#define DHID   128
#define DOUTF  64

typedef __attribute__((ext_vector_type(16))) _Float16 v16h;
typedef __attribute__((ext_vector_type(8)))  _Float16 v8h;
typedef __attribute__((ext_vector_type(8)))  float    v8f;

// ---------------------------------------------------------------- utilities
__global__ void k_zero(float* __restrict__ p, long n) {
    long t = (long)blockIdx.x * blockDim.x + threadIdx.x;
    if (t < n) p[t] = 0.0f;
}

// degree: cnt[dst] += 1 per edge
__global__ void k_deg(const int* __restrict__ dst, float* __restrict__ cnt, int E) {
    int e = blockIdx.x * blockDim.x + threadIdx.x;
    if (e < E) atomicAdd(&cnt[dst[e]], 1.0f);
}

// dis = rsqrt(cnt+1), invcnt = 1/max(cnt,1)
__global__ void k_disinv(const float* __restrict__ cnt, float* __restrict__ dis,
                         float* __restrict__ invcnt, int n) {
    int i = blockIdx.x * blockDim.x + threadIdx.x;
    if (i < n) {
        float c = cnt[i];
        dis[i]    = rsqrtf(c + 1.0f);
        invcnt[i] = 1.0f / fmaxf(c, 1.0f);
    }
}

// ---------------------------------------------------------------- WMMA GEMM
// Y[nrows][ODIM] = X[nrows][KDIM] @ W[KDIM][ODIM], nrows % 16 == 0.
// Block = 256 threads (8 waves). Whole W staged column-major in LDS as f16.
// Each wave: one 16-row tile; A fragments (all KDIM/32 chunks) loaded once to
// registers, then swept across all ODIM/16 column tiles from LDS.
template <int KDIM, int ODIM>
__global__ void k_gemm_wmma(const float* __restrict__ X, const float* __restrict__ W,
                            float* __restrict__ Y, int nrows) {
    constexpr int NCHUNK = KDIM / 32;   // K chunks of 32
    constexpr int NCT    = ODIM / 16;   // 16-wide column tiles
    __shared__ _Float16 wsmT[ODIM * KDIM];      // wsmT[col*KDIM + k] = W[k][col]

    // cooperative stage: global-coalesced reads (consecutive cols), f32->f16
    for (int idx = threadIdx.x; idx < KDIM * ODIM; idx += 256) {
        int k   = idx / ODIM;
        int col = idx - k * ODIM;
        wsmT[col * KDIM + k] = (_Float16)W[idx];
    }
    __syncthreads();

    const int lane = threadIdx.x & 31;
    const int wave = threadIdx.x >> 5;
    const int rowTile = blockIdx.x * 8 + wave;
    if (rowTile >= (nrows >> 4)) return;        // wave-uniform exit

    const int m    = lane & 15;                 // row in tile (A) / col in tile (B,C)
    const int half = lane >> 4;                 // K-half select
    const float* __restrict__ xrow = X + (long)(rowTile * 16 + m) * KDIM;

    // ---- load all A fragments once (two 8-float runs per chunk -> b128 loads)
    v16h afr[NCHUNK];
#pragma unroll
    for (int c = 0; c < NCHUNK; ++c) {
        const int kb = c * 32 + half * 8;
        const float4 q0 = *(const float4*)(xrow + kb);
        const float4 q1 = *(const float4*)(xrow + kb + 4);
        const float4 q2 = *(const float4*)(xrow + kb + 16);
        const float4 q3 = *(const float4*)(xrow + kb + 20);
        afr[c][0]  = (_Float16)q0.x;  afr[c][1]  = (_Float16)q0.y;
        afr[c][2]  = (_Float16)q0.z;  afr[c][3]  = (_Float16)q0.w;
        afr[c][4]  = (_Float16)q1.x;  afr[c][5]  = (_Float16)q1.y;
        afr[c][6]  = (_Float16)q1.z;  afr[c][7]  = (_Float16)q1.w;
        afr[c][8]  = (_Float16)q2.x;  afr[c][9]  = (_Float16)q2.y;
        afr[c][10] = (_Float16)q2.z;  afr[c][11] = (_Float16)q2.w;
        afr[c][12] = (_Float16)q3.x;  afr[c][13] = (_Float16)q3.y;
        afr[c][14] = (_Float16)q3.z;  afr[c][15] = (_Float16)q3.w;
    }

    // ---- sweep column tiles, B fragments from LDS (two ds_load_b128 per chunk)
#pragma unroll
    for (int ct = 0; ct < NCT; ++ct) {
        const _Float16* __restrict__ wcol = wsmT + (ct * 16 + m) * KDIM;
        v8f acc = {};
#pragma unroll
        for (int c = 0; c < NCHUNK; ++c) {
            const int kb = c * 32 + half * 8;
            const v8h blo = *(const v8h*)(wcol + kb);
            const v8h bhi = *(const v8h*)(wcol + kb + 16);
            v16h b;
#pragma unroll
            for (int i = 0; i < 8; ++i) { b[i] = blo[i]; b[8 + i] = bhi[i]; }
            acc = __builtin_amdgcn_wmma_f32_16x16x32_f16(
                      false, afr[c], false, b, (short)0, acc, false, false);
        }
#pragma unroll
        for (int r = 0; r < 8; ++r) {
            int row = rowTile * 16 + half * 8 + r;
            Y[(long)row * ODIM + ct * 16 + m] = acc[r];
        }
    }
}

// ---------------------------------------------------------------- edge passes
// agg[dst] += dis[src]*dis[dst] * xw[src]  (1<<gshift lanes per edge, float4 per lane-group)
__global__ void k_edge_agg(const int* __restrict__ src, const int* __restrict__ dst,
                           const float* __restrict__ dis, const float* __restrict__ xw,
                           float* __restrict__ agg, int gshift, int E) {
    long t = (long)blockIdx.x * blockDim.x + threadIdx.x;
    long e = t >> gshift;
    int  g = (int)(t & ((1 << gshift) - 1));
    if (e >= E) return;
    int s = src[e], d = dst[e];
    float nrm = dis[s] * dis[d];
    int odim = 4 << gshift;
    const float4 v = *((const float4*)(xw + (long)s * odim) + g);
    float* ap = agg + (long)d * odim + g * 4;
    atomicAdd(ap + 0, nrm * v.x);
    atomicAdd(ap + 1, nrm * v.y);
    atomicAdd(ap + 2, nrm * v.z);
    atomicAdd(ap + 3, nrm * v.w);
}

// h = agg + dis^2 * xw + bias
__global__ void k_node_h(const float* __restrict__ agg, const float* __restrict__ xw,
                         const float* __restrict__ dis, const float* __restrict__ bias,
                         float* __restrict__ h, int oshift, long total) {
    long t = (long)blockIdx.x * blockDim.x + threadIdx.x;
    if (t >= total) return;
    long i = t >> oshift;
    int  f = (int)(t & ((1 << oshift) - 1));
    float d2 = dis[i] * dis[i];
    h[t] = agg[t] + d2 * xw[t] + bias[f];
}

// out2[dst] += h[src] + relu(attr*ew + eb)
__global__ void k_edge_msg(const int* __restrict__ src, const int* __restrict__ dst,
                           const float* __restrict__ attr, const float* __restrict__ ew,
                           const float* __restrict__ eb, const float* __restrict__ h,
                           float* __restrict__ out2, int gshift, int E) {
    long t = (long)blockIdx.x * blockDim.x + threadIdx.x;
    long e = t >> gshift;
    int  g = (int)(t & ((1 << gshift) - 1));
    if (e >= E) return;
    int s = src[e], d = dst[e];
    float a = attr[e];
    int odim = 4 << gshift;
    int f = g * 4;
    const float4 hv = *((const float4*)(h + (long)s * odim) + g);
    float m0 = hv.x + fmaxf(a * ew[f + 0] + eb[f + 0], 0.0f);
    float m1 = hv.y + fmaxf(a * ew[f + 1] + eb[f + 1], 0.0f);
    float m2 = hv.z + fmaxf(a * ew[f + 2] + eb[f + 2], 0.0f);
    float m3 = hv.w + fmaxf(a * ew[f + 3] + eb[f + 3], 0.0f);
    float* op = out2 + (long)d * odim + f;
    atomicAdd(op + 0, m0);
    atomicAdd(op + 1, m1);
    atomicAdd(op + 2, m2);
    atomicAdd(op + 3, m3);
}

// x = LN(relu(out2 * invcnt)) * g + b    (feature dim == 128, one wave32 per node)
__global__ void k_relu_ln(const float* __restrict__ out2, const float* __restrict__ invcnt,
                          const float* __restrict__ g, const float* __restrict__ b,
                          float* __restrict__ xo, int n) {
    int node = blockIdx.x * 8 + (threadIdx.x >> 5);
    if (node >= n) return;
    int lane = threadIdx.x & 31;
    float ic = invcnt[node];
    const float4 v4 = *((const float4*)(out2 + (long)node * 128) + lane);
    float v0 = fmaxf(v4.x * ic, 0.0f);
    float v1 = fmaxf(v4.y * ic, 0.0f);
    float v2 = fmaxf(v4.z * ic, 0.0f);
    float v3 = fmaxf(v4.w * ic, 0.0f);
    float s1 = v0 + v1 + v2 + v3;
    float s2 = v0 * v0 + v1 * v1 + v2 * v2 + v3 * v3;
#pragma unroll
    for (int off = 16; off >= 1; off >>= 1) {
        s1 += __shfl_xor(s1, off, 32);
        s2 += __shfl_xor(s2, off, 32);
    }
    float mean = s1 * (1.0f / 128.0f);
    float var  = s2 * (1.0f / 128.0f) - mean * mean;
    float r    = rsqrtf(var + 1e-5f);
    int f = lane * 4;
    float4 o;
    o.x = (v0 - mean) * r * g[f + 0] + b[f + 0];
    o.y = (v1 - mean) * r * g[f + 1] + b[f + 1];
    o.z = (v2 - mean) * r * g[f + 2] + b[f + 2];
    o.w = (v3 - mean) * r * g[f + 3] + b[f + 3];
    *((float4*)(xo + (long)node * 128) + lane) = o;
}

// final layer output: h = out2 * invcnt
__global__ void k_scale(const float* __restrict__ out2, const float* __restrict__ invcnt,
                        float* __restrict__ ho, int oshift, long total) {
    long t = (long)blockIdx.x * blockDim.x + threadIdx.x;
    if (t >= total) return;
    ho[t] = out2[t] * invcnt[t >> oshift];
}

// ---------------------------------------------------------------- pooling
__global__ void k_gcnt(const int* __restrict__ batch, float* __restrict__ gcnt, int n) {
    int i = blockIdx.x * blockDim.x + threadIdx.x;
    if (i < n) atomicAdd(&gcnt[batch[i]], 1.0f);
}

__global__ void k_pool_acc(const int* __restrict__ batch, const float* __restrict__ h,
                           float* __restrict__ pooled, int n) {
    long t = (long)blockIdx.x * blockDim.x + threadIdx.x;   // n * 16 threads (64/4 groups)
    long i = t >> 4;
    int  g = (int)(t & 15);
    if (i >= n) return;
    int bg = batch[i];
    const float4 v = *((const float4*)(h + i * 64) + g);
    float* pp = pooled + (long)bg * 64 + g * 4;
    atomicAdd(pp + 0, v.x);
    atomicAdd(pp + 1, v.y);
    atomicAdd(pp + 2, v.z);
    atomicAdd(pp + 3, v.w);
}

__global__ void k_pool_div(const float* __restrict__ gcnt, float* __restrict__ pooled, long total) {
    long t = (long)blockIdx.x * blockDim.x + threadIdx.x;
    if (t >= total) return;
    pooled[t] /= fmaxf(gcnt[t >> 6], 1.0f);
}

// ---------------------------------------------------------------- launch
static inline int cdiv(long a, int b) { return (int)((a + b - 1) / b); }

extern "C" void kernel_launch(void* const* d_in, const int* in_sizes, int n_in,
                              void* d_out, int out_size, void* d_ws, size_t ws_size,
                              hipStream_t stream) {
    (void)in_sizes; (void)n_in; (void)out_size; (void)ws_size;
    const float* x    = (const float*)d_in[0];
    const int*   ei   = (const int*)d_in[1];
    const float* attr = (const float*)d_in[2];
    const int*   bat  = (const int*)d_in[3];
    const int* src = ei;
    const int* dst = ei + NEDGES;

    const float* gw[3] = { (const float*)d_in[4],  (const float*)d_in[10], (const float*)d_in[16] };
    const float* gb[3] = { (const float*)d_in[5],  (const float*)d_in[11], (const float*)d_in[17] };
    const float* ew[3] = { (const float*)d_in[6],  (const float*)d_in[12], (const float*)d_in[18] };
    const float* eb[3] = { (const float*)d_in[7],  (const float*)d_in[13], (const float*)d_in[19] };
    const float* lg[2] = { (const float*)d_in[8],  (const float*)d_in[14] };
    const float* lb[2] = { (const float*)d_in[9],  (const float*)d_in[15] };

    float* hout   = (float*)d_out;                 // [N, 64]
    float* pooled = hout + (long)NNODES * DOUTF;   // [G, 64]

    // workspace carve-up (floats)
    float* ws   = (float*)d_ws;
    float* cnt  = ws;                       ws += NNODES;
    float* dis  = ws;                       ws += NNODES;
    float* icnt = ws;                       ws += NNODES;
    float* gcnt = ws;                       ws += NGRAPH;
    float* xw   = ws;                       ws += (long)NNODES * DHID;
    float* agg  = ws;                       ws += (long)NNODES * DHID;
    float* hbuf = ws;                       ws += (long)NNODES * DHID;
    float* out2 = ws;                       ws += (long)NNODES * DHID;
    float* xcur = ws;                       ws += (long)NNODES * DHID;

    const int B = 256;
    const int GEMM_BLKS = cdiv(NNODES / 16, 8);    // 8 row tiles per block

    // ---- degree / normalization factors (shared by all layers)
    k_zero<<<cdiv(NNODES, B), B, 0, stream>>>(cnt, NNODES);
    k_deg<<<cdiv(NEDGES, B), B, 0, stream>>>(dst, cnt, NEDGES);
    k_disinv<<<cdiv(NNODES, B), B, 0, stream>>>(cnt, dis, icnt, NNODES);

    // ---- one MPNN layer (post-GEMM part; the GEMM itself is launched by caller)
    auto layer_rest = [&](int odim, int L, float* xnext, bool final_layer) {
        const int gshift = (odim == 128) ? 5 : 4;   // float4 groups per edge
        const int oshift = (odim == 128) ? 7 : 6;
        const long tot   = (long)NNODES * odim;

        k_zero<<<cdiv(tot, B), B, 0, stream>>>(agg, tot);
        k_edge_agg<<<cdiv((long)NEDGES << gshift, B), B, 0, stream>>>(src, dst, dis, xw, agg, gshift, NEDGES);
        k_node_h<<<cdiv(tot, B), B, 0, stream>>>(agg, xw, dis, gb[L], hbuf, oshift, tot);

        k_zero<<<cdiv(tot, B), B, 0, stream>>>(out2, tot);
        k_edge_msg<<<cdiv((long)NEDGES << gshift, B), B, 0, stream>>>(src, dst, attr, ew[L], eb[L], hbuf, out2, gshift, NEDGES);

        if (final_layer) {
            k_scale<<<cdiv(tot, B), B, 0, stream>>>(out2, icnt, xnext, oshift, tot);
        } else {
            k_relu_ln<<<cdiv(NNODES, 8), B, 0, stream>>>(out2, icnt, lg[L], lb[L], xnext, NNODES);
        }
    };

    // layer 0: 128 -> 128
    k_gemm_wmma<DIN, DHID><<<GEMM_BLKS, B, 0, stream>>>(x, gw[0], xw, NNODES);
    layer_rest(DHID, 0, xcur, false);
    // layer 1: 128 -> 128
    k_gemm_wmma<DHID, DHID><<<GEMM_BLKS, B, 0, stream>>>(xcur, gw[1], xw, NNODES);
    layer_rest(DHID, 1, xcur, false);
    // layer 2: 128 -> 64 (final)
    k_gemm_wmma<DHID, DOUTF><<<GEMM_BLKS, B, 0, stream>>>(xcur, gw[2], xw, NNODES);
    layer_rest(DOUTF, 2, hout, true);

    // ---- global mean pool
    k_zero<<<cdiv(NGRAPH * DOUTF, B), B, 0, stream>>>(pooled, NGRAPH * DOUTF);
    k_zero<<<1, B, 0, stream>>>(gcnt, NGRAPH);
    k_gcnt<<<cdiv(NNODES, B), B, 0, stream>>>(bat, gcnt, NNODES);
    k_pool_acc<<<cdiv((long)NNODES * 16, B), B, 0, stream>>>(bat, hout, pooled, NNODES);
    k_pool_div<<<cdiv(NGRAPH * DOUTF, B), B, 0, stream>>>(gcnt, pooled, NGRAPH * DOUTF);
}
